// SparseSimilarityOld_80135499809326
// MI455X (gfx1250) — matrix operand order repaired and checked by reading, not compiled, not run on hardware
//
#include <hip/hip_runtime.h>
#include <hip/hip_bf16.h>

#define NX     16384
#define NYTOT  16384
#define CDIM   128
#define KTOP   15
#define NSPLIT 8
#define STRIP  (NYTOT / NSPLIT)   /* 2048 */
#define NTILES (STRIP / 16)       /* 128  */
#define INV_TAU 5.0f              /* 1/0.2 */

typedef __bf16 bf16x16 __attribute__((ext_vector_type(16)));
typedef __bf16 bf16x8  __attribute__((ext_vector_type(8)));
typedef float  f32x8   __attribute__((ext_vector_type(8)));

// ---------------------------------------------------------------------------
// Kernel 1: fp32 -> bf16 (RNE) conversion of both feature matrices into d_ws.
// ---------------------------------------------------------------------------
__global__ void SparseSim_cvt_bf16(const float* __restrict__ x,
                                   const float* __restrict__ y,
                                   unsigned short* __restrict__ xb,
                                   unsigned short* __restrict__ yb, int n) {
    int i = blockIdx.x * blockDim.x + threadIdx.x;
    if (i < n) {
        unsigned int ux = __float_as_uint(x[i]);
        unsigned int uy = __float_as_uint(y[i]);
        xb[i] = (unsigned short)((ux + 0x7FFFu + ((ux >> 16) & 1u)) >> 16);
        yb[i] = (unsigned short)((uy + 0x7FFFu + ((uy >> 16) & 1u)) >> 16);
    }
}

// ---------------------------------------------------------------------------
// Kernel 2: fused bf16-WMMA GEMM + streaming per-row top-15 + softmax.
// Block = 256 threads (8 wave32). Block b owns rows [16b, 16b+16).
// Wave w streams column strip [w*2048, (w+1)*2048) with double-buffered
// register B-fragments (software pipelined) and keeps a sorted 15-entry
// (val,idx) list per row in LDS, gated by per-row thresholds held in VGPRs.
// After a barrier the block merges the 8 partial lists per row, softmaxes
// the final 15, and writes output.
// ---------------------------------------------------------------------------
__global__ __launch_bounds__(256, 1)
void SparseSim_topk_gemm(const unsigned short* __restrict__ xb,
                         const unsigned short* __restrict__ yb,
                         float* __restrict__ outv,
                         int* __restrict__ outi) {
    __shared__ float svals[NSPLIT][16][16];   // [strip][row][slot]; slot 15 = pad
    __shared__ int   sidx [NSPLIT][16][16];

    const int lane = threadIdx.x & 31;
    const int w    = threadIdx.x >> 5;        // wave id == strip id
    const int rt   = blockIdx.x;              // 16-row tile id

    // --- init wave-private top-k lists (no barrier needed yet) ---
    for (int t = lane; t < 256; t += 32) {
        svals[w][t >> 4][t & 15] = -INFINITY;
        sidx [w][t >> 4][t & 15] = 0;
    }

    const int m    = lane & 15;               // row-in-tile (A) / col-in-tile (B)
    const int half = lane >> 4;

    // --- A fragments: rows rt*16+m, all K=128, resident in VGPRs ---
    // bf16 A k order used here (contiguous, same as B): lane<16 holds K = kb+{0..15};
    // lanes 16..31 hold K = kb+16+{0..15} (half*16 shift).
    const unsigned short* Xrow = xb + ((size_t)rt * 16 + m) * CDIM;
    bf16x16 afrag[4];
#pragma unroll
    for (int kc = 0; kc < 4; ++kc) {
        const int kb = kc * 32;
        bf16x8 lo = *(const bf16x8*)(Xrow + kb + half * 16);
        bf16x8 hi = *(const bf16x8*)(Xrow + kb + half * 16 + 8);
        afrag[kc] = __builtin_shufflevector(lo, hi,
                        0,1,2,3,4,5,6,7,8,9,10,11,12,13,14,15);
    }

    const int stripBase = w * STRIP;
    // bf16 B k order used here (contiguous, same as A): lane holds N = lane&15, K = (lane>>4)*16 + {0..15}
    // => B[k][n] = Y[n][k]: 16 contiguous bf16 from row (colBase+m) of Y.
    const unsigned short* Ybase =
        yb + ((size_t)(stripBase + m)) * CDIM + half * 16;

    // per-row top-15 thresholds in VGPRs (row = (half<<3)+r for slot r)
    float thr[8];
#pragma unroll
    for (int r = 0; r < 8; ++r) thr[r] = -INFINITY;

    auto loadB = [&](int nt, bf16x8* buf) {
        const unsigned short* p = Ybase + (size_t)nt * 16 * CDIM;
#pragma unroll
        for (int kc = 0; kc < 4; ++kc) {
            buf[2 * kc]     = *(const bf16x8*)(p + kc * 32);
            buf[2 * kc + 1] = *(const bf16x8*)(p + kc * 32 + 8);
        }
    };

    auto process = [&](int nt, const bf16x8* buf) {
        const int colBase = stripBase + nt * 16;
        f32x8 c = {};
#pragma unroll
        for (int kc = 0; kc < 4; ++kc) {
            bf16x16 bfrag = __builtin_shufflevector(buf[2 * kc], buf[2 * kc + 1],
                                0,1,2,3,4,5,6,7,8,9,10,11,12,13,14,15);
            c = __builtin_amdgcn_wmma_f32_16x16x32_bf16(
                    false, afrag[kc], false, bfrag, (short)0, c, false, false);
        }
        // D layout: c[r] on lane l = sim[row=(l>>4)*8 + r][col=colBase+(l&15)]
#pragma unroll
        for (int r = 0; r < 8; ++r) {
            const float v = c[r] * INV_TAU;
            unsigned int mask = __builtin_amdgcn_ballot_w32(v > thr[r]);
            while (mask) {
                const int ls = (int)__builtin_ctz(mask);
                mask &= mask - 1;
                const float cv    = __shfl(v, ls, 32);     // candidate value
                const int   ci    = colBase + (ls & 15);   // candidate column
                const int   shalf = ls >> 4;
                const int   tr    = (shalf << 3) + r;      // candidate row
                // 15-lane parallel sorted-shift insert (no-op if cv < list[14])
                float nv = 0.0f;
                if (lane < KTOP) {
                    const float a_i = svals[w][tr][lane];
                    const int   x_i = sidx [w][tr][lane];
                    const float a_p = (lane > 0) ? svals[w][tr][lane - 1] : 0.0f;
                    const int   x_p = (lane > 0) ? sidx [w][tr][lane - 1] : 0;
                    const bool keep = (a_i >= cv);
                    const bool here = !keep && ((lane == 0) || (a_p >= cv));
                    nv = keep ? a_i : (here ? cv : a_p);
                    const int nx = keep ? x_i : (here ? ci : x_p);
                    svals[w][tr][lane] = nv;
                    sidx [w][tr][lane] = nx;
                }
                // lane 14 holds the new 15th-best -> refresh register threshold
                const float nth = __shfl(nv, KTOP - 1, 32);
                if (half == shalf) thr[r] = nth;
            }
        }
    };

    // --- software-pipelined main loop: prefetch B of tile nt+1 / nt+2 ---
    bf16x8 bufA[8], bufB[8];
    loadB(0, bufA);
    for (int nt = 0; nt < NTILES; nt += 2) {
        loadB(nt + 1, bufB);
        process(nt, bufA);
        loadB((nt + 2 < NTILES) ? nt + 2 : nt, bufA);  // clamped: no branch
        process(nt + 1, bufB);
    }

    __syncthreads();

    // --- merge 8 partial lists per row, softmax, write out ---
    // Wave w handles rows 2w, 2w+1. 128 candidate slots (8 strips x 16
    // slots, slot 15 == -inf pad); lane owns slots lane+32j, j=0..3.
#pragma unroll
    for (int rr = 0; rr < 2; ++rr) {
        const int row = 2 * w + rr;
        float cv4[4];
        bool  taken[4];
#pragma unroll
        for (int j = 0; j < 4; ++j) {
            const int cnd = lane + 32 * j;
            cv4[j]   = svals[cnd >> 4][row][cnd & 15];   // slot 15 stays -inf
            taken[j] = false;
        }
        float selv = 0.0f, vmax = 0.0f;
        int   seli = 0;
        for (int it = 0; it < KTOP; ++it) {
            float best  = -INFINITY;
            int   bestc = 0x7fffffff;
#pragma unroll
            for (int j = 0; j < 4; ++j)
                if (!taken[j] && cv4[j] > best) { best = cv4[j]; bestc = lane + 32 * j; }
            // wave-wide argmax (deterministic tie-break on smaller slot id)
            for (int off = 16; off >= 1; off >>= 1) {
                const float ov = __shfl_xor(best, off, 32);
                const int   oc = __shfl_xor(bestc, off, 32);
                if (ov > best || (ov == best && oc < bestc)) { best = ov; bestc = oc; }
            }
            if ((bestc & 31) == lane) taken[bestc >> 5] = true;
            if (it == 0) vmax = best;
            if (lane == it) {
                selv = best;
                seli = sidx[bestc >> 4][row][bestc & 15];
            }
        }
        // softmax over the 15 selected values (lanes 0..14 hold them in order)
        float ev = (lane < KTOP) ? __expf(selv - vmax) : 0.0f;
        float sum = ev;
        for (int off = 16; off >= 1; off >>= 1) sum += __shfl_xor(sum, off, 32);
        if (lane < KTOP) {
            const size_t base = (size_t)(rt * 16 + row) * KTOP;
            outv[base + lane] = ev / sum;
            outi[base + lane] = seli;
        }
    }
}

// ---------------------------------------------------------------------------
extern "C" void kernel_launch(void* const* d_in, const int* in_sizes, int n_in,
                              void* d_out, int out_size, void* d_ws, size_t ws_size,
                              hipStream_t stream) {
    const float* fx = (const float*)d_in[0];   // feat_x [16384,128] fp32
    const float* fy = (const float*)d_in[1];   // feat_y [16384,128] fp32

    unsigned short* xb = (unsigned short*)d_ws;              // 4 MB
    unsigned short* yb = xb + (size_t)NX * CDIM;             // 4 MB

    float* outv = (float*)d_out;                             // [NX, 15] softmax
    int*   outi = (int*)d_out + (size_t)NX * KTOP;           // [NX, 15] indices

    const int n = NX * CDIM;
    SparseSim_cvt_bf16<<<(n + 255) / 256, 256, 0, stream>>>(fx, fy, xb, yb, n);
    SparseSim_topk_gemm<<<NX / 16, 256, 0, stream>>>(xb, yb, outv, outi);
}
